// CausalMaskAttention_53137335386369
// MI455X (gfx1250) — matrix operand order repaired and checked by reading, not compiled, not run on hardware
//
#include <hip/hip_runtime.h>

typedef __attribute__((ext_vector_type(16))) _Float16 v16h;
typedef __attribute__((ext_vector_type(8)))  float    v8f;
typedef __attribute__((ext_vector_type(4)))  int      v4i;

#define BATCH    4
#define SEQ      2048
#define D_MODEL  512
#define N_HEADS  8
#define HEAD_DIM 64
#define THREE_D  1536

// ---------------------------------------------------------------------------
// CDNA5 async global->LDS copy (ASYNCcnt) with synchronous fallback.
// Builtin signature (from clang diagnostics): (v4i as1*, v4i as3*, Ii, Ii).
// ---------------------------------------------------------------------------
__device__ __forceinline__ void async_copy_b128(const void* gsrc, void* ldst) {
#if __has_builtin(__builtin_amdgcn_global_load_async_to_lds_b128)
  __builtin_amdgcn_global_load_async_to_lds_b128(
      (__attribute__((address_space(1))) v4i*)(unsigned long long)gsrc,
      (__attribute__((address_space(3))) v4i*)(unsigned)(unsigned long long)ldst,
      0, 0);
#else
  *(uint4*)ldst = *(const uint4*)gsrc;
#endif
}

__device__ __forceinline__ void wait_async_all() {
#if __has_builtin(__builtin_amdgcn_s_wait_asynccnt)
  __builtin_amdgcn_s_wait_asynccnt(0);
#else
  asm volatile("s_wait_asynccnt 0" ::: "memory");
#endif
}

// ---------------------------------------------------------------------------
// Fragment loaders for V_WMMA_F32_16X16X32_F16 (wave32).
// A (16x32 f16): lane L holds row M = L%16; element e -> K = (e<8?e:e+8)+8*(L/16)
// B (32x16 f16): lane L holds col N = L%16; element e -> K = (e<8?e:e+8)+8*(L/16)
// C/D (16x16 f32): lane L, vgpr r -> row M = r + 8*(L/16), col N = L%16
// Consecutive element pairs have consecutive K, so read as packed dwords
// (the compiler merges these into ds_load_b128).
// ---------------------------------------------------------------------------
__device__ __forceinline__ v16h lds_load_afrag(const _Float16* base, int stride,
                                               int row, int hi) {
  union { v16h v; unsigned u[8]; } r;
  const _Float16* rp = base + row * stride;
#pragma unroll
  for (int j = 0; j < 8; ++j) {
    int e = 2 * j;
    int k = ((e < 8) ? e : e + 8) + 8 * hi;
    r.u[j] = *(const unsigned*)(rp + k);
  }
  return r.v;
}

__device__ __forceinline__ v16h lds_load_bfrag(const _Float16* base, int stride,
                                               int col, int hi) {
  union { v16h v; unsigned u[8]; } r;
  const _Float16* rp = base + col * stride;
#pragma unroll
  for (int j = 0; j < 8; ++j) {
    int k = ((2 * j < 8) ? 2 * j : 2 * j + 8) + 8 * hi;
    r.u[j] = *(const unsigned*)(rp + k);
  }
  return r.v;
}

__device__ __forceinline__ v8f wmma_f16(v16h a, v16h b, v8f c) {
  return __builtin_amdgcn_wmma_f32_16x16x32_f16(false, a, false, b, (short)0, c,
                                                false, false);
}

// ---------------------------------------------------------------------------
// Kernel 1: qkv = x @ w_qkv + b_qkv, scattered into f16 Q (pre-scaled), K, V
// laid out (B, H, S, hd). M=8192, N=1536, K=512. 128x128 tile / workgroup.
// ---------------------------------------------------------------------------
#define LDA 40
#define LDB 40

__global__ __launch_bounds__(256) void qkv_gemm_kernel(
    const float* __restrict__ x, const float* __restrict__ w,
    const float* __restrict__ bias, _Float16* __restrict__ Qh,
    _Float16* __restrict__ Kh, _Float16* __restrict__ Vh) {
  __shared__ __attribute__((aligned(16))) _Float16 As[128 * LDA];
  __shared__ __attribute__((aligned(16))) _Float16 Bs[128 * LDB];

  const int tid  = threadIdx.x;
  const int wid  = tid >> 5;
  const int lane = tid & 31;
  const int hi   = lane >> 4;
  const int ln   = lane & 15;
  const int m0   = blockIdx.x * 128;
  const int n0   = blockIdx.y * 128;
  const int wM   = (wid >> 2) * 64;  // 2 waves along M
  const int wN   = (wid & 3) * 32;   // 4 waves along N

  v8f acc[4][2];
#pragma unroll
  for (int mi = 0; mi < 4; ++mi)
#pragma unroll
    for (int ni = 0; ni < 2; ++ni) acc[mi][ni] = {};

  for (int k0 = 0; k0 < D_MODEL; k0 += 32) {
    __syncthreads();
#pragma unroll
    for (int it = 0; it < 16; ++it) {  // A tile 128x32 (f32 -> f16)
      int idx = it * 256 + tid;
      int kk = idx & 31, mm = idx >> 5;
      As[mm * LDA + kk] = (_Float16)x[(size_t)(m0 + mm) * D_MODEL + k0 + kk];
    }
#pragma unroll
    for (int it = 0; it < 16; ++it) {  // B tile stored transposed Bs[n][k]
      int idx = it * 256 + tid;
      int nn = idx & 127, kk = idx >> 7;
      Bs[nn * LDB + kk] = (_Float16)w[(size_t)(k0 + kk) * THREE_D + n0 + nn];
    }
    __syncthreads();

    v16h bf[2];
#pragma unroll
    for (int ni = 0; ni < 2; ++ni)
      bf[ni] = lds_load_bfrag(Bs, LDB, wN + ni * 16 + ln, hi);
#pragma unroll
    for (int mi = 0; mi < 4; ++mi) {
      v16h af = lds_load_afrag(As, LDA, wM + mi * 16 + ln, hi);
#pragma unroll
      for (int ni = 0; ni < 2; ++ni)
        acc[mi][ni] = wmma_f16(af, bf[ni], acc[mi][ni]);
    }
  }

  const float qscale = 0.125f;  // 1/sqrt(64)
#pragma unroll
  for (int mi = 0; mi < 4; ++mi)
#pragma unroll
    for (int ni = 0; ni < 2; ++ni) {
      union { v8f v; float f[8]; } a;
      a.v = acc[mi][ni];
#pragma unroll
      for (int r = 0; r < 8; ++r) {
        int m = m0 + wM + mi * 16 + r + 8 * hi;
        int n = n0 + wN + ni * 16 + ln;
        float v = a.f[r] + bias[n];
        int part = n >> 9;          // 0=q 1=k 2=v
        int rem  = n & 511;
        int head = rem >> 6, dcol = rem & 63;
        int bb = m >> 11, ss = m & 2047;
        size_t off =
            (((size_t)bb * N_HEADS + head) * SEQ + ss) * HEAD_DIM + dcol;
        if (part == 0)      Qh[off] = (_Float16)(v * qscale);
        else if (part == 1) Kh[off] = (_Float16)v;
        else                Vh[off] = (_Float16)v;
      }
    }
}

// ---------------------------------------------------------------------------
// Kernel 2: flash attention per (b, h, 64-query block). 4 waves, 16 q rows
// per wave. Online causal softmax; QK^T and PV via WMMA. Q/K tiles stream in
// via GLOBAL_LOAD_ASYNC_TO_LDS_B128 (overlapped with the V transpose).
// ---------------------------------------------------------------------------
#define LQ 72  // row stride in halves (144 B, multiple of 16 B)

__global__ __launch_bounds__(128) void attn_kernel(
    const _Float16* __restrict__ Qh, const _Float16* __restrict__ Kh,
    const _Float16* __restrict__ Vh, _Float16* __restrict__ Og) {
  __shared__ __attribute__((aligned(16))) _Float16 Qs[64 * LQ];
  __shared__ __attribute__((aligned(16))) _Float16 Ks[64 * LQ];
  __shared__ __attribute__((aligned(16))) _Float16 Vts[64 * LQ];  // [d][key]
  __shared__ __attribute__((aligned(16))) _Float16 Ps[4 * 16 * LQ];

  const int tid  = threadIdx.x;
  const int w    = tid >> 5;
  const int lane = tid & 31;
  const int hi   = lane >> 4;
  const int ln   = lane & 15;
  const int qb   = blockIdx.x;
  const int b    = blockIdx.y >> 3;
  const int h    = blockIdx.y & 7;
  const size_t head_base = ((size_t)b * N_HEADS + h) * SEQ * HEAD_DIM;

  // Async-load the 64x64 f16 Q tile: 512 x 16-byte chunks.
  {
    const _Float16* gq = Qh + head_base + (size_t)qb * 64 * HEAD_DIM;
#pragma unroll
    for (int it = 0; it < 4; ++it) {
      int idx = it * 128 + tid;        // [0, 512): 8 halves each
      int row = idx >> 3, ch = idx & 7;
      async_copy_b128(gq + idx * 8, &Qs[row * LQ + ch * 8]);
    }
  }
  wait_async_all();
  __syncthreads();

  v16h aq0 = lds_load_afrag(Qs, LQ, w * 16 + ln, hi);       // d 0..31
  v16h aq1 = lds_load_afrag(Qs + 32, LQ, w * 16 + ln, hi);  // d 32..63

  v8f acc[4];
#pragma unroll
  for (int df = 0; df < 4; ++df) acc[df] = {};
  float mrow[8], lrow[8];
#pragma unroll
  for (int r = 0; r < 8; ++r) { mrow[r] = -3.0e38f; lrow[r] = 0.0f; }
  const int q0w = qb * 64 + w * 16;

  for (int kb = 0; kb <= qb; ++kb) {
    __syncthreads();
    const _Float16* gk = Kh + head_base + (size_t)kb * 64 * HEAD_DIM;
    const unsigned* gv =
        (const unsigned*)(Vh + head_base + (size_t)kb * 64 * HEAD_DIM);

    // Issue the K-tile DMA, then overlap with the V transpose.
#pragma unroll
    for (int it = 0; it < 4; ++it) {
      int idx = it * 128 + tid;
      int row = idx >> 3, ch = idx & 7;
      async_copy_b128(gk + idx * 8, &Ks[row * LQ + ch * 8]);
    }
#pragma unroll
    for (int it = 0; it < 16; ++it) {
      int idx = it * 128 + tid;
      int row = idx >> 5, dp = idx & 31;
      union { unsigned u; _Float16 hh[2]; } cv;
      cv.u = gv[idx];
      Vts[(dp * 2 + 0) * LQ + row] = cv.hh[0];
      Vts[(dp * 2 + 1) * LQ + row] = cv.hh[1];
    }
    wait_async_all();
    __syncthreads();

    // S = Q @ K^T  (scale pre-folded into Q)
    union { v8f v; float f[8]; } sf[4];
#pragma unroll
    for (int nf = 0; nf < 4; ++nf) {
      v16h bk0 = lds_load_bfrag(Ks, LQ, nf * 16 + ln, hi);
      v16h bk1 = lds_load_bfrag(Ks + 32, LQ, nf * 16 + ln, hi);
      v8f s = {};
      s = wmma_f16(aq0, bk0, s);
      s = wmma_f16(aq1, bk1, s);
      sf[nf].v = s;
    }

    if (kb == qb) {  // causal mask on the diagonal block
#pragma unroll
      for (int nf = 0; nf < 4; ++nf) {
        int kidx = kb * 64 + nf * 16 + ln;
#pragma unroll
        for (int r = 0; r < 8; ++r) {
          int qidx = q0w + r + 8 * hi;
          if (kidx > qidx) sf[nf].f[r] = -3.0e38f;
        }
      }
    }

    // Row max over 64 keys: 4 frags then xor-reduce within 16-lane halves.
    float rmax[8];
#pragma unroll
    for (int r = 0; r < 8; ++r) {
      float m = sf[0].f[r];
#pragma unroll
      for (int nf = 1; nf < 4; ++nf) m = fmaxf(m, sf[nf].f[r]);
      rmax[r] = m;
    }
#pragma unroll
    for (int msk = 1; msk <= 8; msk <<= 1)
#pragma unroll
      for (int r = 0; r < 8; ++r)
        rmax[r] = fmaxf(rmax[r], __shfl_xor(rmax[r], msk, 32));

    float corr[8], rsum[8];
#pragma unroll
    for (int r = 0; r < 8; ++r) {
      float mn = fmaxf(mrow[r], rmax[r]);
      corr[r] = __expf(mrow[r] - mn);
      mrow[r] = mn;
      rsum[r] = 0.0f;
    }
#pragma unroll
    for (int nf = 0; nf < 4; ++nf)
#pragma unroll
      for (int r = 0; r < 8; ++r) {
        float p = __expf(sf[nf].f[r] - mrow[r]);
        sf[nf].f[r] = p;
        rsum[r] += p;
      }
#pragma unroll
    for (int msk = 1; msk <= 8; msk <<= 1)
#pragma unroll
      for (int r = 0; r < 8; ++r) rsum[r] += __shfl_xor(rsum[r], msk, 32);
#pragma unroll
    for (int r = 0; r < 8; ++r) lrow[r] = lrow[r] * corr[r] + rsum[r];
#pragma unroll
    for (int df = 0; df < 4; ++df) {
      union { v8f v; float f[8]; } t;
      t.v = acc[df];
#pragma unroll
      for (int r = 0; r < 8; ++r) t.f[r] *= corr[r];
      acc[df] = t.v;
    }

    // D-frag layout -> A-frag layout via per-wave LDS round trip.
    __syncthreads();
    _Float16* pw = Ps + w * 16 * LQ;
#pragma unroll
    for (int nf = 0; nf < 4; ++nf)
#pragma unroll
      for (int r = 0; r < 8; ++r)
        pw[(r + 8 * hi) * LQ + nf * 16 + ln] = (_Float16)sf[nf].f[r];
    __syncthreads();
    v16h ap0 = lds_load_afrag(pw, LQ, ln, hi);       // keys 0..31
    v16h ap1 = lds_load_afrag(pw + 32, LQ, ln, hi);  // keys 32..63

#pragma unroll
    for (int df = 0; df < 4; ++df) {
      v16h bv0 = lds_load_bfrag(Vts, LQ, df * 16 + ln, hi);
      v16h bv1 = lds_load_bfrag(Vts + 32, LQ, df * 16 + ln, hi);
      acc[df] = wmma_f16(ap0, bv0, acc[df]);
      acc[df] = wmma_f16(ap1, bv1, acc[df]);
    }
  }

  float inv[8];
#pragma unroll
  for (int r = 0; r < 8; ++r) inv[r] = 1.0f / lrow[r];
#pragma unroll
  for (int df = 0; df < 4; ++df) {
    union { v8f v; float f[8]; } t;
    t.v = acc[df];
#pragma unroll
    for (int r = 0; r < 8; ++r) {
      int s = q0w + r + 8 * hi;
      size_t off =
          ((size_t)b * SEQ + s) * D_MODEL + h * HEAD_DIM + df * 16 + ln;
      Og[off] = (_Float16)(t.f[r] * inv[r]);
    }
  }
}

// ---------------------------------------------------------------------------
// Kernel 3: out = attn_out @ w_out + b_out (fp32 result). M=8192,N=512,K=512.
// A tile is already f16 -> async DMA straight into LDS.
// ---------------------------------------------------------------------------
__global__ __launch_bounds__(256) void out_gemm_kernel(
    const _Float16* __restrict__ Og, const float* __restrict__ w,
    const float* __restrict__ bias, float* __restrict__ out) {
  __shared__ __attribute__((aligned(16))) _Float16 As[128 * LDA];
  __shared__ __attribute__((aligned(16))) _Float16 Bs[128 * LDB];

  const int tid  = threadIdx.x;
  const int wid  = tid >> 5;
  const int lane = tid & 31;
  const int hi   = lane >> 4;
  const int ln   = lane & 15;
  const int m0   = blockIdx.x * 128;
  const int n0   = blockIdx.y * 128;
  const int wM   = (wid >> 2) * 64;
  const int wN   = (wid & 3) * 32;

  v8f acc[4][2];
#pragma unroll
  for (int mi = 0; mi < 4; ++mi)
#pragma unroll
    for (int ni = 0; ni < 2; ++ni) acc[mi][ni] = {};

  for (int k0 = 0; k0 < D_MODEL; k0 += 32) {
    __syncthreads();
    // A tile 128x32 f16: async DMA, 512 x 16-byte chunks (LDA row = 80 B).
#pragma unroll
    for (int it = 0; it < 2; ++it) {
      int idx = it * 256 + tid;        // [0, 512)
      int mm = idx >> 2, ch = idx & 3;
      async_copy_b128(Og + (size_t)(m0 + mm) * D_MODEL + k0 + ch * 8,
                      &As[mm * LDA + ch * 8]);
    }
#pragma unroll
    for (int it = 0; it < 16; ++it) {  // B tile (f32 -> f16), transposed
      int idx = it * 256 + tid;
      int nn = idx & 127, kk = idx >> 7;
      Bs[nn * LDB + kk] = (_Float16)w[(size_t)(k0 + kk) * D_MODEL + n0 + nn];
    }
    wait_async_all();
    __syncthreads();

    v16h bf[2];
#pragma unroll
    for (int ni = 0; ni < 2; ++ni)
      bf[ni] = lds_load_bfrag(Bs, LDB, wN + ni * 16 + ln, hi);
#pragma unroll
    for (int mi = 0; mi < 4; ++mi) {
      v16h af = lds_load_afrag(As, LDA, wM + mi * 16 + ln, hi);
#pragma unroll
      for (int ni = 0; ni < 2; ++ni)
        acc[mi][ni] = wmma_f16(af, bf[ni], acc[mi][ni]);
    }
  }

#pragma unroll
  for (int mi = 0; mi < 4; ++mi)
#pragma unroll
    for (int ni = 0; ni < 2; ++ni) {
      union { v8f v; float f[8]; } a;
      a.v = acc[mi][ni];
#pragma unroll
      for (int r = 0; r < 8; ++r) {
        int m = m0 + wM + mi * 16 + r + 8 * hi;
        int n = n0 + wN + ni * 16 + ln;
        out[(size_t)m * D_MODEL + n] = a.f[r] + bias[n];
      }
    }
}

// ---------------------------------------------------------------------------
extern "C" void kernel_launch(void* const* d_in, const int* in_sizes, int n_in,
                              void* d_out, int out_size, void* d_ws,
                              size_t ws_size, hipStream_t stream) {
  const float* x     = (const float*)d_in[0];
  const float* w_qkv = (const float*)d_in[1];
  const float* b_qkv = (const float*)d_in[2];
  const float* w_out = (const float*)d_in[3];
  const float* b_out = (const float*)d_in[4];
  float* out = (float*)d_out;

  char* ws = (char*)d_ws;
  const size_t seg =
      (size_t)BATCH * N_HEADS * SEQ * HEAD_DIM * sizeof(_Float16);  // 8 MB
  _Float16* Qh = (_Float16*)(ws + 0 * seg);
  _Float16* Kh = (_Float16*)(ws + 1 * seg);
  _Float16* Vh = (_Float16*)(ws + 2 * seg);
  _Float16* Og = (_Float16*)(ws + 3 * seg);

  dim3 g1(64, 12);  // 8192/128 x 1536/128
  qkv_gemm_kernel<<<g1, dim3(256), 0, stream>>>(x, w_qkv, b_qkv, Qh, Kh, Vh);

  dim3 g2(32, 32);  // 2048/64 query blocks x (B*H)
  attn_kernel<<<g2, dim3(128), 0, stream>>>(Qh, Kh, Vh, Og);

  dim3 g3(64, 4);   // 8192/128 x 512/128
  out_gemm_kernel<<<g3, dim3(256), 0, stream>>>(Og, w_out, b_out, out);
}
